// ContrastiveLoss_5660766896848
// MI455X (gfx1250) — matrix operand order repaired and checked by reading, not compiled, not run on hardware
//
#include <hip/hip_runtime.h>
#include <hip/hip_fp16.h>

typedef __attribute__((ext_vector_type(16))) _Float16 v16h;
typedef __attribute__((ext_vector_type(8)))  float    v8f;

#define LAYERS   4
#define NHALF    2048        // N = B*K
#define NROWS    4096        // 2N rows per layer
#define DIM      128
#define TEMP_INV 5.0f        // 1/0.2
#define CTILE    128         // columns staged in LDS per iteration
#define NCT      (NROWS / CTILE)

union Frag16 { uint4 u[2]; v16h h; };
union Pack4h { _Float16 h[4]; uint2 u2; };

__device__ __forceinline__ float wave_sum32(float v) {
    v += __shfl_xor(v, 1);
    v += __shfl_xor(v, 2);
    v += __shfl_xor(v, 4);
    v += __shfl_xor(v, 8);
    v += __shfl_xor(v, 16);
    return v;
}

// ---------------------------------------------------------------------------
// Kernel 1: L2-normalize emb_i / emb_j rows (f32), emit f16 Z and f32 pos.
// One wave (32 lanes) per sample index n; each lane owns 4 of 128 dims.
// ---------------------------------------------------------------------------
__global__ void k_normalize(const float* __restrict__ emb_i,
                            const float* __restrict__ emb_j,
                            _Float16* __restrict__ Z,
                            float* __restrict__ pos)
{
    const int wid   = (blockIdx.x * blockDim.x + threadIdx.x) >> 5;
    const int lane  = threadIdx.x & 31;
    const int layer = wid >> 11;            // / NHALF
    const int n     = wid & (NHALF - 1);

    const size_t base = ((size_t)layer * NHALF + n) * DIM + lane * 4;
    const float4 vi = *(const float4*)(emb_i + base);
    const float4 vj = *(const float4*)(emb_j + base);

    float si = wave_sum32(vi.x*vi.x + vi.y*vi.y + vi.z*vi.z + vi.w*vi.w);
    float sj = wave_sum32(vj.x*vj.x + vj.y*vj.y + vj.z*vj.z + vj.w*vj.w);
    const float inv_i = 1.0f / fmaxf(sqrtf(si), 1e-12f);
    const float inv_j = 1.0f / fmaxf(sqrtf(sj), 1e-12f);

    const float4 zi = make_float4(vi.x*inv_i, vi.y*inv_i, vi.z*inv_i, vi.w*inv_i);
    const float4 zj = make_float4(vj.x*inv_j, vj.y*inv_j, vj.z*inv_j, vj.w*inv_j);

    // positive pair similarity in full f32
    float p = wave_sum32(zi.x*zj.x + zi.y*zj.y + zi.z*zj.z + zi.w*zj.w);

    _Float16* Zl = Z + (size_t)layer * NROWS * DIM;
    Pack4h pi_, pj_;
    pi_.h[0] = (_Float16)zi.x; pi_.h[1] = (_Float16)zi.y;
    pi_.h[2] = (_Float16)zi.z; pi_.h[3] = (_Float16)zi.w;
    pj_.h[0] = (_Float16)zj.x; pj_.h[1] = (_Float16)zj.y;
    pj_.h[2] = (_Float16)zj.z; pj_.h[3] = (_Float16)zj.w;
    *(uint2*)(Zl + (size_t)n * DIM + lane * 4)            = pi_.u2;
    *(uint2*)(Zl + (size_t)(NHALF + n) * DIM + lane * 4)  = pj_.u2;

    if (lane == 0) pos[layer * NHALF + n] = p;
}

// ---------------------------------------------------------------------------
// Kernel 2: fused Z·Zᵀ (WMMA f16 -> f32) + exp/row-sum + log -> loss partial.
// Grid: LAYERS * (NROWS/128) blocks of 256 threads (8 waves).
// Each wave owns 16 rows; A fragments stay in registers across all columns.
// Column tiles are staged with CDNA5 async global->LDS DMA (ASYNCcnt).
// ---------------------------------------------------------------------------
__global__ void k_simloss(const _Float16* __restrict__ Z,
                          const float* __restrict__ pos,
                          const float* __restrict__ joint_valid,
                          float* __restrict__ loss_acc)
{
    __shared__ __align__(16) _Float16 tileB[CTILE * DIM];   // 32 KB

    const int layer = blockIdx.x >> 5;
    const int rb    = blockIdx.x & 31;
    const int wave  = threadIdx.x >> 5;
    const int lane  = threadIdx.x & 31;
    const int m0    = rb * 128 + wave * 16;

    const _Float16* Zl = Z + (size_t)layer * NROWS * DIM;

    // ---- A fragments: 16x32 f16 per K-chunk, ISA layout:
    //   lane<16 : M=lane,    K = {c*32+0..7,  c*32+16..23}
    //   lane>=16: M=lane-16, K = {c*32+8..15, c*32+24..31}
    const int mrow  = m0 + (lane & 15);
    const int kbase = (lane < 16) ? 0 : 8;
    Frag16 afrag[4];
    #pragma unroll
    for (int c4 = 0; c4 < 4; ++c4) {
        const _Float16* ap = Zl + (size_t)mrow * DIM + c4 * 32 + kbase;
        afrag[c4].u[0] = *(const uint4*)(ap);
        afrag[c4].u[1] = *(const uint4*)(ap + 16);
    }

    const int rbase = (lane < 16) ? 0 : 8;   // C/D row offset
    const int nlane = lane & 15;             // C/D column within tile
    const int kb2   = (lane < 16) ? 0 : 8;   // B K-offset within chunk

    // per-thread staging stripe: 128 B = 8 x b128 async DMAs
    const int srow  = threadIdx.x >> 1;
    const int spart = threadIdx.x & 1;
    const unsigned lds_dst =
        (unsigned)(uintptr_t)(tileB + srow * DIM + spart * 64);

    float acc[8];
    #pragma unroll
    for (int r = 0; r < 8; ++r) acc[r] = 0.0f;

    for (int ct = 0; ct < NCT; ++ct) {
        // ---- async stage of 128 columns (rows of Z) into LDS, no VGPR bounce
        {
            const unsigned long long gsrc = (unsigned long long)(uintptr_t)
                (Zl + (size_t)(ct * CTILE + srow) * DIM + spart * 64);
            #pragma unroll
            for (int i = 0; i < 8; ++i) {
                // INST_OFFSET is added to BOTH the LDS and global address
                asm volatile("global_load_async_to_lds_b128 %0, %1, off offset:%2"
                             :: "v"(lds_dst), "v"(gsrc), "i"(i * 16)
                             : "memory");
            }
            if (ct + 1 < NCT)   // CDNA5 global_prefetch_b8 for next tile
                __builtin_prefetch(Zl + (size_t)((ct + 1) * CTILE + srow) * DIM
                                      + spart * 64, 0, 0);
            asm volatile("s_wait_asynccnt 0x0" ::: "memory");
        }
        __syncthreads();

        // ---- 8 sub-tiles of 16 columns, each: 4 accumulating WMMAs over K=128
        for (int n0 = 0; n0 < 8; ++n0) {
            v8f c = {};
            #pragma unroll
            for (int c4 = 0; c4 < 4; ++c4) {
                Frag16 b;
                const uint4* bs =
                    (const uint4*)(tileB + (n0 * 16 + nlane) * DIM + c4 * 32 + kb2);
                b.u[0] = bs[0];
                b.u[1] = bs[2];
                c = __builtin_amdgcn_wmma_f32_16x16x32_f16(
                        false, afrag[c4].h, false, b.h, (short)0, c, false, false);
            }
            const int col = ct * CTILE + n0 * 16 + nlane;
            #pragma unroll
            for (int r = 0; r < 8; ++r) {
                const int row = m0 + rbase + r;
                acc[r] += (row == col) ? 0.0f : __expf(c[r] * TEMP_INV);
            }
        }
        __syncthreads();
    }

    // ---- per-row: denom = rowsum (diag already excluded); loss partial
    float partial = 0.0f;
    #pragma unroll
    for (int r = 0; r < 8; ++r) {
        float t = acc[r];
        t += __shfl_xor(t, 1);
        t += __shfl_xor(t, 2);
        t += __shfl_xor(t, 4);
        t += __shfl_xor(t, 8);
        if (nlane == 0) {
            const int row = m0 + rbase + r;
            const int pi  = (row < NHALF) ? row : row - NHALF;
            partial += joint_valid[pi] *
                       (logf(t) - pos[layer * NHALF + pi] * TEMP_INV);
        }
    }
    if (nlane == 0) atomicAdd(loss_acc, partial);
}

// ---------------------------------------------------------------------------
// Kernel 3: finalize — all_num = sum(joint_valid); out = loss / (2*all_num)
// ---------------------------------------------------------------------------
__global__ void k_finalize(const float* __restrict__ joint_valid,
                           const float* __restrict__ loss_acc,
                           float* __restrict__ out)
{
    __shared__ float sh[256];
    float s = 0.0f;
    for (int i = threadIdx.x; i < NHALF; i += 256) s += joint_valid[i];
    sh[threadIdx.x] = s;
    __syncthreads();
    for (int off = 128; off > 0; off >>= 1) {
        if (threadIdx.x < off) sh[threadIdx.x] += sh[threadIdx.x + off];
        __syncthreads();
    }
    if (threadIdx.x == 0) out[0] = loss_acc[0] / (2.0f * sh[0]);
}

// ---------------------------------------------------------------------------
extern "C" void kernel_launch(void* const* d_in, const int* in_sizes, int n_in,
                              void* d_out, int out_size, void* d_ws, size_t ws_size,
                              hipStream_t stream)
{
    const float* emb_i       = (const float*)d_in[0];
    const float* emb_j       = (const float*)d_in[1];
    const float* joint_valid = (const float*)d_in[2];
    float* out = (float*)d_out;

    // Workspace layout
    const size_t Z_OFF   = 0;
    const size_t Z_BYTES = (size_t)LAYERS * NROWS * DIM * sizeof(_Float16); // 4 MB
    const size_t POS_OFF = Z_OFF + Z_BYTES;
    const size_t POS_BYTES = (size_t)LAYERS * NHALF * sizeof(float);        // 32 KB
    const size_t ACC_OFF = POS_OFF + POS_BYTES;

    char* ws = (char*)d_ws;
    _Float16* Z        = (_Float16*)(ws + Z_OFF);
    float*    pos      = (float*)(ws + POS_OFF);
    float*    loss_acc = (float*)(ws + ACC_OFF);

    hipMemsetAsync(loss_acc, 0, sizeof(float), stream);

    // Kernel 1: LAYERS*NHALF waves, 8 waves per block
    k_normalize<<<(LAYERS * NHALF) / 8, 256, 0, stream>>>(emb_i, emb_j, Z, pos);

    // Kernel 2: LAYERS * 32 row-blocks
    k_simloss<<<LAYERS * (NROWS / 128), 256, 0, stream>>>(Z, pos, joint_valid, loss_acc);

    // Kernel 3: finalize scalar
    k_finalize<<<1, 256, 0, stream>>>(joint_valid, loss_acc, out);
}
